// GCNEdgeClassifier_84945863180466
// MI455X (gfx1250) — matrix operand order repaired and checked by reading, not compiled, not run on hardware
//
#include <hip/hip_runtime.h>
#include <hip/hip_bf16.h>

typedef _Float16 half8  __attribute__((ext_vector_type(8)));
typedef _Float16 half16 __attribute__((ext_vector_type(16)));
typedef float    f32x8  __attribute__((ext_vector_type(8)));

// ---------------------------------------------------------------------------
// A-fragment loader for v_wmma_f32_16x16x32_f16.
// A is 16(M) x 32(K) f16. K order used here (contiguous, same as B):
//   lanes 0-15 : M=lane,    K = {0..15}   (kb = 0)
//   lanes 16-31: M=lane-16, K = {16..31}  (kb = 8)
// p points at the start of this row's 32-half K-slice.
// ---------------------------------------------------------------------------
__device__ __forceinline__ half16 load_a_frag(const _Float16* p, int kb) {
    half8 lo = *reinterpret_cast<const half8*>(p + 2 * kb);        // K = 2kb+0..7
    half8 hi = *reinterpret_cast<const half8*>(p + 2 * kb + 8);   // K = 2kb+8..15
    return __builtin_shufflevector(lo, hi, 0, 1, 2, 3, 4, 5, 6, 7,
                                           8, 9, 10, 11, 12, 13, 14, 15);
}

// ---------------------------------------------------------------------------
// Small prep kernels
// ---------------------------------------------------------------------------
__global__ void k_deg(const long long* __restrict__ col, float* __restrict__ deg, int E) {
    int i = blockIdx.x * blockDim.x + threadIdx.x;
    if (i < E) atomicAdd(&deg[col[i]], 1.0f);
}

__global__ void k_dinv(const float* __restrict__ deg, float* __restrict__ dinv, int N) {
    int i = blockIdx.x * blockDim.x + threadIdx.x;
    if (i < N) dinv[i] = rsqrtf(deg[i] + 1.0f);   // +1 self-loop; always > 0
}

__global__ void k_cvt_weights(const float* __restrict__ W2, const float* __restrict__ W3,
                              _Float16* __restrict__ W2f, _Float16* __restrict__ W3f) {
    int i = blockIdx.x * blockDim.x + threadIdx.x;
    if (i < 64 * 64) W2f[i] = (_Float16)W2[i];
    int j = i - 64 * 64;
    if (j >= 0 && j < 128 * 64) W3f[j] = (_Float16)W3[j];
}

// hw1 = x @ W1  (N x 3)@(3 x 64); fused self-loop init m1 = dinv^2 * hw1
__global__ void k_hw1(const float* __restrict__ x, const float* __restrict__ W1,
                      const float* __restrict__ dinv,
                      float* __restrict__ hw, float* __restrict__ m, int N) {
    int i = blockIdx.x * blockDim.x + threadIdx.x;
    if (i >= N * 64) return;
    int n = i >> 6, j = i & 63;
    const float* xr = x + n * 3;
    float v = xr[0] * W1[j] + xr[1] * W1[64 + j] + xr[2] * W1[128 + j];
    float d = dinv[n];
    hw[i] = v;
    m[i] = d * d * v;
}

// m[col] += dinv[row]*dinv[col] * hw[row]   per edge, 4 feats per thread
__global__ void k_agg(const long long* __restrict__ row, const long long* __restrict__ col,
                      const float* __restrict__ dinv, const float* __restrict__ hw,
                      float* __restrict__ m, int E) {
    int i = blockIdx.x * blockDim.x + threadIdx.x;
    if (i >= E * 16) return;
    int e = i >> 4, g = i & 15;
    long long r = row[e], c = col[e];
    float s = dinv[r] * dinv[c];
    float4 v = *(reinterpret_cast<const float4*>(hw + (size_t)r * 64) + g);
    float* mp = m + (size_t)c * 64 + g * 4;
    atomicAdd(mp + 0, s * v.x);
    atomicAdd(mp + 1, s * v.y);
    atomicAdd(mp + 2, s * v.z);
    atomicAdd(mp + 3, s * v.w);
}

// h = f16(relu(m + b))
__global__ void k_bias_relu_h16(const float* __restrict__ m, const float* __restrict__ b,
                                _Float16* __restrict__ h, int N) {
    int i = blockIdx.x * blockDim.x + threadIdx.x;
    if (i < N * 64) h[i] = (_Float16)fmaxf(m[i] + b[i & 63], 0.0f);
}

// ---------------------------------------------------------------------------
// GEMM2: hw = h1f16 @ W2f16 -> f32  ([N,64]@[64,64]) via WMMA.
// Fused epilogue also writes m2 = dinv^2 * hw (self-loop init for layer 2).
// B pre-swizzled in LDS, then held register-resident across the tile loop.
// ---------------------------------------------------------------------------
__global__ __launch_bounds__(256) void k_gemm64(const _Float16* __restrict__ hin,
                                                const _Float16* __restrict__ wf16,
                                                const float* __restrict__ dinv,
                                                float* __restrict__ hw,
                                                float* __restrict__ m, int N) {
    __shared__ _Float16 ldsB[8 * 32 * 16];   // [frag][lane][i]  8 KB
    const int tid = threadIdx.x;
    for (int idx = tid; idx < 8 * 32 * 16; idx += 256) {
        int i = idx & 15, lane = (idx >> 4) & 31, frag = idx >> 9;
        int ks = frag >> 2, nt = frag & 3;
        int k = ks * 32 + ((lane & 16) ? 16 : 0) + i;
        int n = nt * 16 + (lane & 15);
        ldsB[idx] = wf16[k * 64 + n];
    }
    __syncthreads();

    const int lane = tid & 31, wave = tid >> 5;
    const int nl = lane & 15;
    const int kb = (lane & 16) ? 8 : 0;
    const int Mb = (lane & 16) ? 8 : 0;

    half16 breg[8];                           // register-resident B (64 VGPRs)
    #pragma unroll
    for (int f = 0; f < 8; ++f)
        breg[f] = *reinterpret_cast<const half16*>(&ldsB[f * 512 + lane * 16]);

    const int numTiles = (N + 15) >> 4;
    for (int tile = blockIdx.x * 8 + wave; tile < numTiles; tile += gridDim.x * 8) {
        const int nbase = tile << 4;
        int node = nbase + nl;
        if (node >= N) node = N - 1;
        const _Float16* hp = hin + (size_t)node * 64;

        f32x8 acc[4] = {};
        #pragma unroll
        for (int ks = 0; ks < 2; ++ks) {
            half16 a = load_a_frag(hp + ks * 32, kb);
            #pragma unroll
            for (int nt = 0; nt < 4; ++nt)
                acc[nt] = __builtin_amdgcn_wmma_f32_16x16x32_f16(
                    false, a, false, breg[ks * 4 + nt], (short)0, acc[nt], false, false);
        }
        if (nbase + 16 <= N) {                   // wave-uniform fast path
            #pragma unroll
            for (int r = 0; r < 8; ++r) {
                int rw = nbase + Mb + r;
                float d = dinv[rw];
                float ss = d * d;
                #pragma unroll
                for (int nt = 0; nt < 4; ++nt) {
                    float v = acc[nt][r];
                    size_t idx = (size_t)rw * 64 + nt * 16 + nl;
                    hw[idx] = v;
                    m[idx]  = ss * v;
                }
            }
        } else {
            #pragma unroll
            for (int r = 0; r < 8; ++r) {
                int rw = nbase + Mb + r;
                if (rw < N) {
                    float d = dinv[rw];
                    float ss = d * d;
                    #pragma unroll
                    for (int nt = 0; nt < 4; ++nt) {
                        float v = acc[nt][r];
                        size_t idx = (size_t)rw * 64 + nt * 16 + nl;
                        hw[idx] = v;
                        m[idx]  = ss * v;
                    }
                }
            }
        }
    }
}

// ---------------------------------------------------------------------------
// Edge MLP (dominant, ~52.8 GFLOP):
//   z   = relu([h[row]|h[col]] @ W3 + b3)   -- WMMA f16, K=128, 16 wmma/tile
//   out = z @ W4 + b4                       -- VALU + butterfly fold
// One wave = 16 edges. B (W3) is register-resident (16 frags = 128 VGPRs);
// inner loop = 8 global_load_b128 (A gathers) + 16 v_wmma + epilogue.
// Butterfly fold: 15 shuffles leave lane l of each half owning row l&7.
// ---------------------------------------------------------------------------
__global__ __launch_bounds__(256) void k_edge_mlp(
    const long long* __restrict__ row, const long long* __restrict__ col,
    const _Float16* __restrict__ h2, const _Float16* __restrict__ w3f16,
    const float* __restrict__ b3, const float* __restrict__ W4,
    const float* __restrict__ b4, float* __restrict__ out, int E) {

    __shared__ _Float16 ldsB[16 * 32 * 16];  // [frag][lane][i]  16 KB
    const int tid = threadIdx.x;
    for (int idx = tid; idx < 16 * 32 * 16; idx += 256) {
        int i = idx & 15, lane = (idx >> 4) & 31, frag = idx >> 9;
        int ks = frag >> 2, nt = frag & 3;
        int k = ks * 32 + ((lane & 16) ? 16 : 0) + i;   // B layout: K contiguous per half-wave
        int n = nt * 16 + (lane & 15);
        ldsB[idx] = w3f16[k * 64 + n];
    }
    __syncthreads();

    const int lane = tid & 31, wave = tid >> 5;
    const int nl = lane & 15;
    const int kb = (lane & 16) ? 8 : 0;
    const int Mb = (lane & 16) ? 8 : 0;

    half16 breg[16];                          // register-resident B (128 VGPRs)
    #pragma unroll
    for (int f = 0; f < 16; ++f)
        breg[f] = *reinterpret_cast<const half16*>(&ldsB[f * 512 + lane * 16]);

    float b3v[4], w4v[4];
    #pragma unroll
    for (int nt = 0; nt < 4; ++nt) {
        b3v[nt] = b3[nt * 16 + nl];
        w4v[nt] = W4[nt * 16 + nl];
    }
    const float b4v = b4[0];

    const int numTiles = (E + 15) >> 4;
    for (int tile = blockIdx.x * 8 + wave; tile < numTiles; tile += gridDim.x * 8) {
        const int ebase = tile << 4;
        int e = ebase + nl;
        int ec = (e < E) ? e : (E - 1);          // clamp, keep EXEC full for WMMA
        const long long rn = row[ec], cn = col[ec];
        const _Float16* hr = h2 + (size_t)rn * 64;
        const _Float16* hc = h2 + (size_t)cn * 64;

        f32x8 acc[4] = {};
        #pragma unroll
        for (int ks = 0; ks < 4; ++ks) {         // K slices 0,32 from h[row]; 64,96 from h[col]
            const _Float16* hp = (ks < 2) ? hr : hc;
            half16 a = load_a_frag(hp + (ks & 1) * 32, kb);
            #pragma unroll
            for (int nt = 0; nt < 4; ++nt)
                acc[nt] = __builtin_amdgcn_wmma_f32_16x16x32_f16(
                    false, a, false, breg[ks * 4 + nt], (short)0, acc[nt], false, false);
        }

        // Layer 2 per-element: z = relu(acc + b3), partial dot with W4.
        float a8[8];
        #pragma unroll
        for (int r = 0; r < 8; ++r) {
            float sr = 0.0f;
            #pragma unroll
            for (int nt = 0; nt < 4; ++nt) {
                float z = acc[nt][r] + b3v[nt];
                z = fmaxf(z, 0.0f);
                sr += z * w4v[nt];
            }
            a8[r] = sr;
        }

        // Butterfly fold across the 16-lane half: 8+4+2+1 shuffles.
        #pragma unroll
        for (int r = 0; r < 8; ++r) a8[r] += __shfl_xor(a8[r], 8, 32);

        const bool hi4 = (lane & 4) != 0;
        float k4[4];
        #pragma unroll
        for (int r = 0; r < 4; ++r) {
            float snd = hi4 ? a8[r] : a8[r + 4];
            float kp  = hi4 ? a8[r + 4] : a8[r];
            k4[r] = kp + __shfl_xor(snd, 4, 32);
        }
        const bool hi2 = (lane & 2) != 0;
        float k2[2];
        #pragma unroll
        for (int r = 0; r < 2; ++r) {
            float snd = hi2 ? k4[r] : k4[r + 2];
            float kp  = hi2 ? k4[r + 2] : k4[r];
            k2[r] = kp + __shfl_xor(snd, 2, 32);
        }
        const bool hi1 = (lane & 1) != 0;
        float snd = hi1 ? k2[0] : k2[1];
        float kp  = hi1 ? k2[1] : k2[0];
        float tot = kp + __shfl_xor(snd, 1, 32);   // lane l owns row l&7

        int eo = ebase + Mb + nl;                  // nl<8: row nl of this half
        if (nl < 8 && eo < E) out[eo] = tot + b4v;
    }
}

// ---------------------------------------------------------------------------
// Launch
// ---------------------------------------------------------------------------
extern "C" void kernel_launch(void* const* d_in, const int* in_sizes, int n_in,
                              void* d_out, int out_size, void* d_ws, size_t ws_size,
                              hipStream_t stream) {
    (void)n_in; (void)out_size; (void)ws_size;
    const float*     x   = (const float*)d_in[0];
    const long long* ei  = (const long long*)d_in[1];   // int64 edge_index [2,E]
    const float*     W1  = (const float*)d_in[2];
    const float*     b1  = (const float*)d_in[3];
    const float*     W2  = (const float*)d_in[4];
    const float*     b2  = (const float*)d_in[5];
    const float*     W3  = (const float*)d_in[6];
    const float*     b3  = (const float*)d_in[7];
    const float*     W4  = (const float*)d_in[8];
    const float*     b4  = (const float*)d_in[9];
    float* out = (float*)d_out;

    const int N = in_sizes[0] / 3;
    const int E = in_sizes[1] / 2;
    const long long* row = ei;
    const long long* colp = ei + E;

    // workspace layout (256B aligned)
    char* ws = (char*)d_ws;
    size_t off = 0;
    auto alloc = [&](size_t bytes) -> void* {
        void* p = ws + off;
        off = (off + bytes + 255) & ~(size_t)255;
        return p;
    };
    float*    deg  = (float*)alloc((size_t)N * 4);
    float*    dinv = (float*)alloc((size_t)N * 4);
    float*    bufA = (float*)alloc((size_t)N * 64 * 4);   // hw1, later hw2
    float*    bufB = (float*)alloc((size_t)N * 64 * 4);   // m1, later m2
    _Float16* h1f  = (_Float16*)alloc((size_t)N * 64 * 2);
    _Float16* h2f  = (_Float16*)alloc((size_t)N * 64 * 2);
    _Float16* W2f  = (_Float16*)alloc(64 * 64 * 2);
    _Float16* W3f  = (_Float16*)alloc(128 * 64 * 2);

    const int T = 256;
    const int NE = (E + T - 1) / T;
    const int NN = (N + T - 1) / T;
    const int NF = (N * 64 + T - 1) / T;         // N*64 = 6.4M, fits int
    const int NA = (E * 16 + T - 1) / T;         // E*16 = 51.2M, fits int

    hipMemsetAsync(deg, 0, (size_t)N * 4, stream);
    k_deg<<<NE, T, 0, stream>>>(colp, deg, E);
    k_dinv<<<NN, T, 0, stream>>>(deg, dinv, N);
    k_cvt_weights<<<(64 * 64 + 128 * 64 + T - 1) / T, T, 0, stream>>>(W2, W3, W2f, W3f);

    // Layer 1 (self-loop init fused into hw1 producer)
    k_hw1<<<NF, T, 0, stream>>>(x, W1, dinv, bufA, bufB, N);
    k_agg<<<NA, T, 0, stream>>>(row, colp, dinv, bufA, bufB, E);
    k_bias_relu_h16<<<NF, T, 0, stream>>>(bufB, b1, h1f, N);

    // Layer 2 (WMMA GEMM with fused self-loop init, then aggregate)
    k_gemm64<<<784, T, 0, stream>>>(h1f, W2f, dinv, bufA, bufB, N);
    k_agg<<<NA, T, 0, stream>>>(row, colp, dinv, bufA, bufB, E);
    k_bias_relu_h16<<<NF, T, 0, stream>>>(bufB, b2, h2f, N);

    // Edge scorer (dominant WMMA kernel)
    k_edge_mlp<<<1024, T, 0, stream>>>(row, colp, h2f, W3f, b3, W4, b4, out, E);
}
